// VectorQuantizer_68375879352394
// MI455X (gfx1250) — matrix operand order repaired and checked by reading, not compiled, not run on hardware
//
#include <hip/hip_runtime.h>
#include <hip/hip_bf16.h>

typedef __attribute__((ext_vector_type(16))) _Float16 v16h;
typedef __attribute__((ext_vector_type(8)))  float    v8f;
typedef __attribute__((ext_vector_type(4)))  int      v4i;

#ifndef __has_builtin
#define __has_builtin(x) 0
#endif
#if __has_builtin(__builtin_amdgcn_global_load_async_to_lds_b128)
#define USE_ASYNC_LDS 1
#else
#define USE_ASYNC_LDS 0
#endif

#define DIM       256
#define NCODES    1024
#define HW        1024
#define BATCH     64
#define NPIX      (BATCH * HW)          // 65536
#define TOTELEM   (BATCH * DIM * HW)    // 16777216
#define PIX_PER_WG 128
#define KCHUNK    64                    // codes per LDS chunk (double-buffered)
#define CROW      (KCHUNK + 8)          // padded row (halfs): 144B lane stride
#define NCHUNKS   (NCODES / KCHUNK)     // 16
#define KT_PER_CHUNK (KCHUNK / 16)      // 4 k-tiles of 16 codes
#define DCH       (DIM / 32)            // 8 WMMA K-steps to cover D=256

// ---------------------------------------------------------------------------
// Kernel 1: emb fp32 [1024][256] -> embT f16 [d][k] + fp32 norms per code.
// ---------------------------------------------------------------------------
__global__ __launch_bounds__(256)
void vq_prep_kernel(const float* __restrict__ emb,
                    _Float16* __restrict__ embT,
                    float* __restrict__ enorm) {
    int k = blockIdx.x;        // 0..1023
    int d = threadIdx.x;       // 0..255
    float v = emb[k * DIM + d];
    embT[(size_t)d * NCODES + k] = (_Float16)v;

    __shared__ float red[256];
    red[d] = v * v;
    __syncthreads();
    for (int s = 128; s > 0; s >>= 1) {
        if (d < s) red[d] += red[d + s];
        __syncthreads();
    }
    if (d == 0) enorm[k] = red[0];
}

// ---------------------------------------------------------------------------
// Kernel 2: WMMA distance GEMM + running argmin, with double-buffered
// async global->LDS staging of the codebook chunks.
//   dist(n,k) = ||e_k||^2 - 2 * dot(z_n, e_k)
// ---------------------------------------------------------------------------
__global__ __launch_bounds__(256)
void vq_argmin_kernel(const float* __restrict__ z,
                      const _Float16* __restrict__ embT,
                      const float* __restrict__ enorm,
                      float* __restrict__ idx_out) {
    __shared__ _Float16 zA[PIX_PER_WG * DIM];      // 64KB: z slab [m][d]
    __shared__ _Float16 cbuf[2][DIM * CROW];       // 2 x 36KB: emb chunks [d][kk] (padded)

    const int tid  = threadIdx.x;
    const int wave = tid >> 5;
    const int lane = tid & 31;
    const int n0   = blockIdx.x * PIX_PER_WG;      // global pixel base
    const int b    = n0 >> 10;
    const int hw0  = n0 & 1023;

    // ---- kick async DMA of the first two codebook chunks immediately ------
    // (separate LDS region -> overlaps the whole z staging + A build below)
#if USE_ASYNC_LDS
    typedef __attribute__((address_space(1))) v4i* gptr_v4i;
    typedef __attribute__((address_space(3))) v4i* lptr_v4i;
    gptr_v4i gsrc = (gptr_v4i)(void*)embT;         // row = NCODES/8 = 128 v4i
    #define STAGE_CHUNK(CH, SEL)                                               \
        {                                                                      \
            lptr_v4i ldst = (lptr_v4i)(void*)&cbuf[(SEL)][0];                  \
            for (int i = tid; i < DIM * (KCHUNK / 8); i += 256) {              \
                int d = i >> 3;   /* 8 groups of 8 halfs per d-row */          \
                int g = i & 7;                                                 \
                __builtin_amdgcn_global_load_async_to_lds_b128(                \
                    gsrc + (size_t)d * (NCODES / 8) + (CH) * (KCHUNK / 8) + g, \
                    ldst + d * (CROW / 8) + g, 0, 0);                          \
            }                                                                  \
        }
#else
    #define STAGE_CHUNK(CH, SEL)                                               \
        for (int i = tid; i < DIM * (KCHUNK / 8); i += 256) {                  \
            int d = i >> 3;                                                    \
            int g = i & 7;                                                     \
            *(uint4*)&cbuf[(SEL)][d * CROW + g * 8] =                          \
                *(const uint4*)&embT[(size_t)d * NCODES + (CH) * KCHUNK + g * 8]; \
        }
#endif

    STAGE_CHUNK(0, 0);
    STAGE_CHUNK(1, 1);

    // ---- stage z slab (b128 reads, coalesced along hw), cvt to f16 --------
    for (int g = tid; g < DIM * (PIX_PER_WG / 4); g += 256) {
        int d  = g >> 5;           // 0..255
        int m4 = (g & 31) * 4;     // pixel group base
        float4 v = *(const float4*)&z[((size_t)(b * DIM + d) << 10) + hw0 + m4];
        zA[(m4 + 0) * DIM + d] = (_Float16)v.x;
        zA[(m4 + 1) * DIM + d] = (_Float16)v.y;
        zA[(m4 + 2) * DIM + d] = (_Float16)v.z;
        zA[(m4 + 3) * DIM + d] = (_Float16)v.w;
    }
    __syncthreads();

    // ---- build persistent A operand (ISA 16-bit A 16x32 layout) -----------
    v16h A[DCH];
    {
        const int m  = (wave << 4) + (lane & 15);
        const int hb = (lane >> 4) * 8;
        #pragma unroll
        for (int c = 0; c < DCH; ++c) {
            int d0 = 32 * c + hb;
            ((uint4*)&A[c])[0] = *(const uint4*)&zA[m * DIM + d0];
            ((uint4*)&A[c])[1] = *(const uint4*)&zA[m * DIM + d0 + 16];
        }
    }

    float minv[8];
    int   mini[8];
    #pragma unroll
    for (int j = 0; j < 8; ++j) { minv[j] = 3.4e38f; mini[j] = 0; }

    // chunk 0 ready: wait own 8 of 16 outstanding async loads, then barrier
#if USE_ASYNC_LDS
    asm volatile("s_wait_asynccnt 0x8" ::: "memory");
#endif
    __syncthreads();

    // ---- main pipelined loop over codebook chunks -------------------------
    for (int ch = 0; ch < NCHUNKS; ++ch) {
        _Float16* cb = &cbuf[ch & 1][0];

        // batched preload of this chunk's code norms (one per column lane)
        float en[KT_PER_CHUNK];
        #pragma unroll
        for (int kt = 0; kt < KT_PER_CHUNK; ++kt)
            en[kt] = enorm[ch * KCHUNK + kt * 16 + (lane & 15)];

        #pragma unroll
        for (int kt = 0; kt < KT_PER_CHUNK; ++kt) {
            v16h Bv[DCH];
            #pragma unroll
            for (int dc = 0; dc < DCH; ++dc) {
                // B operand (ISA 16-bit B 32x16): lane%16 = code column, elements = d (table order)
                const _Float16* bp = &cb[(32 * dc + (lane >> 4) * 8) * CROW + kt * 16 + (lane & 15)];
                #pragma unroll
                for (int s = 0; s < 16; ++s)
                    Bv[dc][s] = bp[((s & 7) + (s >> 3) * 16) * CROW];
            }
            v8f c = {};
            #pragma unroll
            for (int dc = 0; dc < DCH; ++dc) {
                c = __builtin_amdgcn_wmma_f32_16x16x32_f16(
                        false, A[dc], false, Bv[dc], (short)0, c, false, false);
            }
            int kcol = ch * KCHUNK + kt * 16 + (lane & 15);
            #pragma unroll
            for (int j = 0; j < 8; ++j) {
                float dist = __builtin_fmaf(-2.0f, c[j], en[kt]);
                if (dist < minv[j]) { minv[j] = dist; mini[j] = kcol; }
            }
        }

        if (ch + 1 < NCHUNKS) {
            __syncthreads();                 // all waves done reading buf[ch&1]
            if (ch + 2 < NCHUNKS) {
                STAGE_CHUNK(ch + 2, ch & 1); // refill freed buffer (overlaps next compute)
                __builtin_prefetch(&embT[(size_t)tid * NCODES + (ch + 2) * KCHUNK], 0, 0);
#if USE_ASYNC_LDS
                // 16 outstanding: oldest 8 = chunk ch+1 (in-order completion)
                asm volatile("s_wait_asynccnt 0x8" ::: "memory");
#endif
            } else {
#if USE_ASYNC_LDS
                asm volatile("s_wait_asynccnt 0x0" ::: "memory");
#endif
            }
            __syncthreads();                 // chunk ch+1 visible to all waves
        }
    }

    // ---- cross-lane argmin within each 16-lane half (rows 0-7 / 8-15) -----
    #pragma unroll
    for (int s = 1; s < 16; s <<= 1) {
        #pragma unroll
        for (int j = 0; j < 8; ++j) {
            float ov = __shfl_xor(minv[j], s, 32);
            int   oi = __shfl_xor(mini[j], s, 32);
            if (ov < minv[j] || (ov == minv[j] && oi < mini[j])) {
                minv[j] = ov; mini[j] = oi;
            }
        }
    }
    if ((lane & 15) == 0) {
        int rowBase = (lane >> 4) * 8;   // lane 0 -> rows 0..7, lane 16 -> 8..15
        #pragma unroll
        for (int j = 0; j < 8; ++j) {
            int pix = n0 + (wave << 4) + rowBase + j;
            idx_out[pix] = (float)mini[j];
        }
    }
}

// ---------------------------------------------------------------------------
// Kernel 3: z_q_out = emb[idx] scattered back to (B,D,H,W), plus fixed-order
// per-block partial sums of (z_q - z)^2.
// ---------------------------------------------------------------------------
__global__ __launch_bounds__(256)
void vq_zq_loss_kernel(const float* __restrict__ z,
                       const float* __restrict__ emb,
                       const float* __restrict__ idx_f,
                       float* __restrict__ zq_out,
                       float* __restrict__ partial) {
    float acc = 0.0f;
    const size_t stride = (size_t)gridDim.x * 256;
    for (size_t i = (size_t)blockIdx.x * 256 + threadIdx.x; i < TOTELEM; i += stride) {
        int hw = (int)(i & 1023);
        int d  = (int)((i >> 10) & 255);
        int b  = (int)(i >> 18);
        int idx = (int)idx_f[(b << 10) + hw];
        float e  = emb[idx * DIM + d];
        float zv = z[i];
        zq_out[i] = e;              // straight-through output == z_q numerically
        float df = e - zv;
        acc += df * df;
    }
    __shared__ float red[256];
    red[threadIdx.x] = acc;
    __syncthreads();
    for (int s = 128; s > 0; s >>= 1) {
        if (threadIdx.x < s) red[threadIdx.x] += red[threadIdx.x + s];
        __syncthreads();
    }
    if (threadIdx.x == 0) partial[blockIdx.x] = red[0];
}

// ---------------------------------------------------------------------------
// Kernel 4: deterministic final reduction -> loss = 1.25 * mean((z_q - z)^2)
// ---------------------------------------------------------------------------
__global__ __launch_bounds__(256)
void vq_loss_final_kernel(const float* __restrict__ partial, int n,
                          float* __restrict__ loss_out) {
    __shared__ float red[256];
    float acc = 0.0f;
    for (int i = threadIdx.x; i < n; i += 256) acc += partial[i];
    red[threadIdx.x] = acc;
    __syncthreads();
    for (int s = 128; s > 0; s >>= 1) {
        if (threadIdx.x < s) red[threadIdx.x] += red[threadIdx.x + s];
        __syncthreads();
    }
    if (threadIdx.x == 0)
        *loss_out = 1.25f * red[0] / (float)TOTELEM;
}

// ---------------------------------------------------------------------------
extern "C" void kernel_launch(void* const* d_in, const int* in_sizes, int n_in,
                              void* d_out, int out_size, void* d_ws, size_t ws_size,
                              hipStream_t stream) {
    const float* z   = (const float*)d_in[0];   // (64,256,32,32) fp32
    const float* emb = (const float*)d_in[1];   // (1024,256)     fp32

    float* out   = (float*)d_out;
    float* zq    = out;                         // 16,777,216 floats
    float* loss  = out + (size_t)TOTELEM;       // 1 float
    float* idxf  = out + (size_t)TOTELEM + 1;   // 65,536 floats (indices)

    char* ws = (char*)d_ws;
    _Float16* embT  = (_Float16*)ws;                              // 512 KB
    float*    enorm = (float*)(ws + (size_t)DIM * NCODES * 2);    // 4 KB
    float*    partl = (float*)(ws + (size_t)DIM * NCODES * 2 + NCODES * 4);

    const int ZQ_BLOCKS = 4096;

    vq_prep_kernel<<<NCODES, 256, 0, stream>>>(emb, embT, enorm);
    vq_argmin_kernel<<<NPIX / PIX_PER_WG, 256, 0, stream>>>(z, embT, enorm, idxf);
    vq_zq_loss_kernel<<<ZQ_BLOCKS, 256, 0, stream>>>(z, emb, idxf, zq, partl);
    vq_loss_final_kernel<<<1, 256, 0, stream>>>(partl, ZQ_BLOCKS, loss);
}
